// Qwen2_5OmniToken2Wav_53446573031488
// MI455X (gfx1250) — matrix operand order repaired and checked by reading, not compiled, not run on hardware
//
#include <hip/hip_runtime.h>
#include <hip/hip_bf16.h>

// ---------------------------------------------------------------------------
// Sizes (compile-time constants from the reference)
// ---------------------------------------------------------------------------
#define BATCH     2
#define SEQ       1536
#define DIM       1024
#define QKVDIM    3072   // fused Q|K|V output width
#define HEADS     16
#define HEAD_DIM  64
#define BLK       24     // attention block size
// query tile = 2 blocks = 48 rows; key window = 4 blocks = 96 cols

typedef __attribute__((ext_vector_type(16))) __bf16 v16bf;
typedef __attribute__((ext_vector_type(8)))  float  v8f;

union Frag16 { v16bf v; uint4 q[2]; };
union Bf8    { uint4 q; __bf16 h[8]; };

// ---------------------------------------------------------------------------
// CDNA5 async global->LDS (ASYNCcnt-tracked, no VGPR round trip)
// ---------------------------------------------------------------------------
__device__ __forceinline__ void async_load_b128(unsigned lds_off,
                                                const void* gaddr) {
  asm volatile("global_load_async_to_lds_b128 %0, %1, off"
               :
               : "v"(lds_off), "v"((unsigned long long)gaddr)
               : "memory");
}
__device__ __forceinline__ void wait_async0() {
  asm volatile("s_wait_asynccnt 0" ::: "memory");
}

// ---------------------------------------------------------------------------
// fp32 -> bf16 conversion (8 elements / thread), contiguous
// ---------------------------------------------------------------------------
__global__ void f32_to_bf16_kernel(const float* __restrict__ in,
                                   __bf16* __restrict__ out, int n) {
  int i = (blockIdx.x * blockDim.x + threadIdx.x) * 8;
  if (i < n) {
    float4 a = *reinterpret_cast<const float4*>(in + i);
    float4 b = *reinterpret_cast<const float4*>(in + i + 4);
    Bf8 o;
    o.h[0] = (__bf16)a.x; o.h[1] = (__bf16)a.y;
    o.h[2] = (__bf16)a.z; o.h[3] = (__bf16)a.w;
    o.h[4] = (__bf16)b.x; o.h[5] = (__bf16)b.y;
    o.h[6] = (__bf16)b.z; o.h[7] = (__bf16)b.w;
    *reinterpret_cast<uint4*>(out + i) = o.q;
  }
}

// fp32 [R x 1024] -> bf16 written into [R x out_stride] at column col0
__global__ void f32_to_bf16_stride_kernel(const float* __restrict__ in,
                                          __bf16* __restrict__ out,
                                          int out_stride, int col0, int n) {
  int i = (blockIdx.x * blockDim.x + threadIdx.x) * 8;
  if (i < n) {
    int r = i >> 10, c = i & 1023;  // input row width = DIM = 1024
    float4 a = *reinterpret_cast<const float4*>(in + i);
    float4 b = *reinterpret_cast<const float4*>(in + i + 4);
    Bf8 o;
    o.h[0] = (__bf16)a.x; o.h[1] = (__bf16)a.y;
    o.h[2] = (__bf16)a.z; o.h[3] = (__bf16)a.w;
    o.h[4] = (__bf16)b.x; o.h[5] = (__bf16)b.y;
    o.h[6] = (__bf16)b.z; o.h[7] = (__bf16)b.w;
    *reinterpret_cast<uint4*>(out + (size_t)r * out_stride + col0 + c) = o.q;
  }
}

// concat three fp32 bias vectors of length DIM into one [3*DIM]
__global__ void concat_bias_kernel(const float* __restrict__ b0,
                                   const float* __restrict__ b1,
                                   const float* __restrict__ b2,
                                   float* __restrict__ cb) {
  int n = blockIdx.x * blockDim.x + threadIdx.x;
  if (n < QKVDIM) {
    float v = (n < DIM) ? b0[n] : (n < 2 * DIM ? b1[n - DIM] : b2[n - 2 * DIM]);
    cb[n] = v;
  }
}

// ---------------------------------------------------------------------------
// Tiled bf16 WMMA GEMM:  C[M,N] = A[M,K] @ B[K,N] + bias
//   workgroup tile 128x64, 8 waves (4x2), each wave 32x32 (2x2 wmma tiles)
//   K staged 32 at a time via async global->LDS, double buffered
// ---------------------------------------------------------------------------
#define AST 40   // A LDS row stride: 32 + 8 pad (bf16 units)
#define BST 72   // B LDS row stride: 64 + 8 pad

__device__ __forceinline__ void stage_gemm_tile_async(
    const __bf16* __restrict__ A, const __bf16* __restrict__ Bw, int K, int N,
    int m0g, int n0g, int kt, int t, __bf16* Asl, __bf16* Bsl) {
  // A tile 128x32: 512 x 16B chunks, 2 per thread
#pragma unroll
  for (int c0 = 0; c0 < 2; ++c0) {
    int c = t + c0 * 256;
    int r = c >> 2, cc = c & 3;
    async_load_b128((unsigned)(size_t)&Asl[r * AST + cc * 8],
                    A + (size_t)(m0g + r) * K + kt + cc * 8);
  }
  // B tile 32x64: 256 x 16B chunks, 1 per thread
  {
    int r = t >> 3, cc = t & 7;
    async_load_b128((unsigned)(size_t)&Bsl[r * BST + cc * 8],
                    Bw + (size_t)(kt + r) * N + n0g + cc * 8);
  }
}

template <bool OUT_F32>
__global__ __launch_bounds__(256) void gemm_bf16_wmma(
    const __bf16* __restrict__ A, const __bf16* __restrict__ Bw,
    const float* __restrict__ bias,
    __bf16* __restrict__ Cb, float* __restrict__ Cf,
    int M, int N, int K) {
  __shared__ __align__(16) __bf16 Asl[2][128 * AST];
  __shared__ __align__(16) __bf16 Bsl[2][32 * BST];

  const int t    = threadIdx.x;
  const int lane = t & 31;
  const int wave = t >> 5;
  const int m0g  = blockIdx.y * 128;
  const int n0g  = blockIdx.x * 64;
  const int waveM = (wave >> 1) * 32;  // 0/32/64/96
  const int waveN = (wave & 1) * 32;   // 0/32

  v8f acc[2][2];
#pragma unroll
  for (int i = 0; i < 2; ++i)
#pragma unroll
    for (int j = 0; j < 2; ++j) acc[i][j] = {};

  // prologue: fill buffer 0
  stage_gemm_tile_async(A, Bw, K, N, m0g, n0g, 0, t, Asl[0], Bsl[0]);
  wait_async0();
  __syncthreads();

  int buf = 0;
  for (int kt = 0; kt < K; kt += 32) {
    // issue async fill of the other buffer while we compute
    if (kt + 32 < K)
      stage_gemm_tile_async(A, Bw, K, N, m0g, n0g, kt + 32, t,
                            Asl[buf ^ 1], Bsl[buf ^ 1]);

    // B fragments: lane -> N = lane&15; K 0..7 / 16..23, lanes 16-31 -> +8
    Frag16 bf[2];
#pragma unroll
    for (int nt = 0; nt < 2; ++nt) {
      const __bf16* p = &Bsl[buf][((lane >> 4) << 3) * BST + waveN + nt * 16 + (lane & 15)];
#pragma unroll
      for (int s = 0; s < 16; ++s) bf[nt].v[s] = p[(s < 8 ? s : s + 8) * BST];
    }
    // A fragments: lanes 0-15 -> K 0..7 / 16..23, lanes 16-31 -> +8
    const int kb = (lane >> 4) << 3;
#pragma unroll
    for (int mt = 0; mt < 2; ++mt) {
      Frag16 a;
      const __bf16* p = &Asl[buf][(waveM + mt * 16 + (lane & 15)) * AST + kb];
      a.q[0] = *reinterpret_cast<const uint4*>(p);
      a.q[1] = *reinterpret_cast<const uint4*>(p + 16);
#pragma unroll
      for (int nt = 0; nt < 2; ++nt)
        acc[mt][nt] = __builtin_amdgcn_wmma_f32_16x16x32_bf16(
            false, a.v, false, bf[nt].v, (short)0, acc[mt][nt], false, false);
    }

    wait_async0();      // our async writes into buf^1 have landed in LDS
    __syncthreads();    // everyone done reading buf / writing buf^1
    buf ^= 1;
  }

  // epilogue: C layout = VGPR r -> M = r (+8 for lanes 16-31), N = lane&15
  const int lm = (lane >> 4) << 3;
  const int ln = lane & 15;
#pragma unroll
  for (int mt = 0; mt < 2; ++mt)
#pragma unroll
    for (int nt = 0; nt < 2; ++nt)
#pragma unroll
      for (int r = 0; r < 8; ++r) {
        int m = m0g + waveM + mt * 16 + r + lm;
        int n = n0g + waveN + nt * 16 + ln;
        float v = acc[mt][nt][r] + bias[n];
        if (OUT_F32)
          Cf[(size_t)m * N + n] = v;
        else
          Cb[(size_t)m * N + n] = (__bf16)v;
      }
}

// ---------------------------------------------------------------------------
// Partial rotary: head 0 only, interleaved pairs; ld = row stride (elements)
// ---------------------------------------------------------------------------
__global__ void rope_head0_kernel(__bf16* __restrict__ Q, __bf16* __restrict__ Kc,
                                  const int* __restrict__ pos_ids, int ld) {
  int s = blockIdx.x % SEQ;
  int b = blockIdx.x / SEQ;
  int i = threadIdx.x;  // pair index 0..31
  float pos  = (float)pos_ids[s];
  float freq = __expf(-((float)(2 * i) / (float)HEAD_DIM) * logf(10000.0f));
  float sn, cs;
  __sincosf(pos * freq, &sn, &cs);
  size_t base = ((size_t)(b * SEQ + s)) * ld + 2 * i;  // head 0 = dims 0..63
  float q1 = (float)Q[base], q2 = (float)Q[base + 1];
  Q[base]     = (__bf16)(q1 * cs - q2 * sn);
  Q[base + 1] = (__bf16)(q2 * cs + q1 * sn);
  float k1 = (float)Kc[base], k2 = (float)Kc[base + 1];
  Kc[base]     = (__bf16)(k1 * cs - k2 * sn);
  Kc[base + 1] = (__bf16)(k2 * cs + k1 * sn);
}

// ---------------------------------------------------------------------------
// Block-banded attention, one workgroup per (b, h, 48-query tile)
//   Q tile 48x64, key window 96 (q0-24 .. q0+71), WMMA for QK^T and PV
//   Q/K/V have row stride ld (fused QKV buffer); O has row stride DIM
// ---------------------------------------------------------------------------
__global__ __launch_bounds__(256) void block_attn_kernel(
    const __bf16* __restrict__ Q, const __bf16* __restrict__ Kc,
    const __bf16* __restrict__ V, __bf16* __restrict__ O, int ld) {
  constexpr int QST  = 72;   // 64 + 8 (bf16)
  constexpr int KTST = 104;  // 96 + 8 (bf16), K stored transposed [d][key]
  constexpr int VST  = 72;   // 64 + 8 (bf16)
  constexpr int SST  = 100;  // 96 + 4 (f32)
  constexpr int PST  = 104;  // 96 + 8 (bf16) softmax probs
  __shared__ __align__(16) __bf16 Qs[48 * QST];
  __shared__ __align__(16) __bf16 KTs[HEAD_DIM * KTST];
  __shared__ __align__(16) __bf16 Vs[96 * VST];
  __shared__ __align__(16) float  Ss[48 * SST];
  __shared__ __align__(16) __bf16 Ps[48 * PST];

  const int t = threadIdx.x, lane = t & 31, wave = t >> 5;
  const int qt = blockIdx.x, h = blockIdx.y, b = blockIdx.z;
  const int q0 = qt * 48;       // multiple of 2*BLK
  const int kstart = q0 - BLK;  // fixed 96-wide key window
  const size_t rowbase = (size_t)b * SEQ;
  const size_t hoff = (size_t)h * HEAD_DIM;

  // ---- stage Q (48x64) ----
  for (int c = t; c < 48 * 8; c += 256) {
    int r = c >> 3, ch = c & 7;
    uint4 v = *reinterpret_cast<const uint4*>(
        Q + (rowbase + q0 + r) * ld + hoff + ch * 8);
    *reinterpret_cast<uint4*>(&Qs[r * QST + ch * 8]) = v;
  }
  // ---- stage K transposed: KTs[d][key], zero-fill out-of-range keys ----
  for (int c = t; c < 96 * 8; c += 256) {
    int kk = c >> 3, ch = c & 7;
    int kpos = kstart + kk;
    Bf8 v;
    if (kpos >= 0 && kpos < SEQ)
      v.q = *reinterpret_cast<const uint4*>(
          Kc + (rowbase + kpos) * ld + hoff + ch * 8);
    else
      v.q = make_uint4(0, 0, 0, 0);
#pragma unroll
    for (int j = 0; j < 8; ++j) KTs[(ch * 8 + j) * KTST + kk] = v.h[j];
  }
  // ---- stage V (96x64), zero-fill out-of-range ----
  for (int c = t; c < 96 * 8; c += 256) {
    int kk = c >> 3, ch = c & 7;
    int kpos = kstart + kk;
    uint4 v = (kpos >= 0 && kpos < SEQ)
                  ? *reinterpret_cast<const uint4*>(
                        V + (rowbase + kpos) * ld + hoff + ch * 8)
                  : make_uint4(0, 0, 0, 0);
    *reinterpret_cast<uint4*>(&Vs[kk * VST + ch * 8]) = v;
  }
  __syncthreads();

  // ---- scores: 3x6 tiles of 16x16, contraction K = 64 ----
  for (int tile = wave; tile < 18; tile += 8) {
    int mt = tile / 6, nt = tile % 6;
    v8f acc = {};
#pragma unroll
    for (int kc = 0; kc < HEAD_DIM; kc += 32) {
      Frag16 a, bf;
      const __bf16* ap =
          &Qs[(mt * 16 + (lane & 15)) * QST + kc + ((lane >> 4) << 3)];
      a.q[0] = *reinterpret_cast<const uint4*>(ap);
      a.q[1] = *reinterpret_cast<const uint4*>(ap + 16);
      const __bf16* bp = &KTs[(kc + ((lane >> 4) << 3)) * KTST + nt * 16 + (lane & 15)];
#pragma unroll
      for (int s = 0; s < 16; ++s) bf.v[s] = bp[(s < 8 ? s : s + 8) * KTST];
      acc = __builtin_amdgcn_wmma_f32_16x16x32_bf16(false, a.v, false, bf.v,
                                                    (short)0, acc, false, false);
    }
    // Branch-free, division-free mask. With q0 a multiple of 24 and the
    // window starting at q0-24:  blockdiff = ki/24 - qi/24 - 1, so
    // allowed  <=>  qi/24 <= ki/24 <= qi/24 + 2  (tile-relative compares).
    const int lm = (lane >> 4) << 3, ln = lane & 15;
    const int ki = nt * 16 + ln;          // 0..95
    const int kpos = kstart + ki;
    const bool inr = ((unsigned)kpos) < (unsigned)SEQ;
    const int kd = (ki >= 24) + (ki >= 48) + (ki >= 72);  // ki/24
#pragma unroll
    for (int r = 0; r < 8; ++r) {
      int qi = mt * 16 + r + lm;          // 0..47
      int qd = (qi >= 24);                // qi/24
      bool ok = inr & (kd >= qd) & (kd <= qd + 2);
      Ss[qi * SST + ki] = ok ? acc[r] * 0.125f : -1e30f;
    }
  }
  __syncthreads();

  // ---- softmax over 96 keys: row = t>>2, each quad-lane does 24 cols ----
  if (t < 192) {
    int r = t >> 2, qd = t & 3;
    const float* row = &Ss[r * SST];
    float mx = -1e30f;
#pragma unroll
    for (int j = 0; j < 24; ++j) mx = fmaxf(mx, row[qd * 24 + j]);
    mx = fmaxf(mx, __shfl_xor(mx, 1, 32));
    mx = fmaxf(mx, __shfl_xor(mx, 2, 32));
    float e[24], sum = 0.f;
#pragma unroll
    for (int j = 0; j < 24; ++j) {
      e[j] = __expf(row[qd * 24 + j] - mx);
      sum += e[j];
    }
    sum += __shfl_xor(sum, 1, 32);
    sum += __shfl_xor(sum, 2, 32);
    float inv = 1.0f / sum;
#pragma unroll
    for (int j = 0; j < 24; ++j)
      Ps[r * PST + qd * 24 + j] = (__bf16)(e[j] * inv);
  }
  __syncthreads();

  // ---- attn @ V: 3x4 tiles of 16x16, contraction K = 96 ----
  for (int tile = wave; tile < 12; tile += 8) {
    int mt = tile >> 2, nt = tile & 3;
    v8f acc = {};
#pragma unroll
    for (int kc = 0; kc < 96; kc += 32) {
      Frag16 a, bf;
      const __bf16* ap =
          &Ps[(mt * 16 + (lane & 15)) * PST + kc + ((lane >> 4) << 3)];
      a.q[0] = *reinterpret_cast<const uint4*>(ap);
      a.q[1] = *reinterpret_cast<const uint4*>(ap + 16);
      const __bf16* bp = &Vs[(kc + ((lane >> 4) << 3)) * VST + nt * 16 + (lane & 15)];
#pragma unroll
      for (int s = 0; s < 16; ++s) bf.v[s] = bp[(s < 8 ? s : s + 8) * VST];
      acc = __builtin_amdgcn_wmma_f32_16x16x32_bf16(false, a.v, false, bf.v,
                                                    (short)0, acc, false, false);
    }
    const int lm = (lane >> 4) << 3, ln = lane & 15;
#pragma unroll
    for (int r = 0; r < 8; ++r) {
      int qi = mt * 16 + r + lm;
      int d = nt * 16 + ln;
      O[(rowbase + q0 + qi) * DIM + hoff + d] = (__bf16)acc[r];
    }
  }
}

// ---------------------------------------------------------------------------
// Launch
// ---------------------------------------------------------------------------
extern "C" void kernel_launch(void* const* d_in, const int* in_sizes, int n_in,
                              void* d_out, int out_size, void* d_ws,
                              size_t ws_size, hipStream_t stream) {
  const float* x  = (const float*)d_in[0];
  const float* Wq = (const float*)d_in[1];
  const float* bq = (const float*)d_in[2];
  const float* Wk = (const float*)d_in[3];
  const float* bk = (const float*)d_in[4];
  const float* Wv = (const float*)d_in[5];
  const float* bv = (const float*)d_in[6];
  const float* Wo = (const float*)d_in[7];
  const float* bo = (const float*)d_in[8];
  const int*  pos = (const int*)d_in[9];
  float* out = (float*)d_out;

  const size_t NX  = (size_t)BATCH * SEQ * DIM;  // 3,145,728
  const size_t NW  = (size_t)DIM * DIM;          // 1,048,576
  const size_t M   = (size_t)BATCH * SEQ;        // 3072

  char* w = (char*)d_ws;
  auto alloc = [&](size_t bytes) {
    char* r = w;
    w += (bytes + 255) & ~(size_t)255;
    return r;
  };
  __bf16* xb   = (__bf16*)alloc(NX * 2);            // x in bf16
  __bf16* W3   = (__bf16*)alloc(NW * 3 * 2);        // [1024 x 3072] Wq|Wk|Wv
  __bf16* Wob  = (__bf16*)alloc(NW * 2);            // [1024 x 1024]
  float*  cb   = (float*) alloc(QKVDIM * 4);        // concat bias
  __bf16* QKV  = (__bf16*)alloc(M * QKVDIM * 2);    // [3072 x 3072]
  __bf16* Ob   = (__bf16*)alloc(NX * 2);            // attention output

  const int thr = 256;
  f32_to_bf16_kernel<<<(int)(NX / 8 / thr), thr, 0, stream>>>(x, xb, (int)NX);
  f32_to_bf16_stride_kernel<<<(int)(NW / 8 / thr), thr, 0, stream>>>(
      Wq, W3, QKVDIM, 0, (int)NW);
  f32_to_bf16_stride_kernel<<<(int)(NW / 8 / thr), thr, 0, stream>>>(
      Wk, W3, QKVDIM, DIM, (int)NW);
  f32_to_bf16_stride_kernel<<<(int)(NW / 8 / thr), thr, 0, stream>>>(
      Wv, W3, QKVDIM, 2 * DIM, (int)NW);
  f32_to_bf16_kernel<<<(int)(NW / 8 / thr), thr, 0, stream>>>(Wo, Wob, (int)NW);
  concat_bias_kernel<<<(QKVDIM + thr - 1) / thr, thr, 0, stream>>>(bq, bk, bv, cb);

  // fused QKV projection: [3072 x 1024] @ [1024 x 3072]  -> 1152 workgroups
  gemm_bf16_wmma<false><<<dim3(QKVDIM / 64, M / 128), 256, 0, stream>>>(
      xb, W3, cb, QKV, nullptr, (int)M, QKVDIM, DIM);

  rope_head0_kernel<<<BATCH * SEQ, 32, 0, stream>>>(QKV, QKV + DIM, pos, QKVDIM);

  block_attn_kernel<<<dim3(SEQ / 48, HEADS, BATCH), 256, 0, stream>>>(
      QKV, QKV + DIM, QKV + 2 * DIM, Ob, QKVDIM);

  // output projection -> fp32 d_out
  gemm_bf16_wmma<true><<<dim3(DIM / 64, M / 128), 256, 0, stream>>>(
      Ob, Wob, bo, nullptr, out, (int)M, DIM, DIM);
}
